// deform_block_head_22797686407586
// MI455X (gfx1250) — matrix operand order repaired and checked by reading, not compiled, not run on hardware
//
// CDNA5 / gfx1250 (MI455X) implementation of the DGCNN-style edge-conv block.
// Big GEMMs use v_wmma_f32_16x16x32_bf16 (wave32, A-fragments pre-swizzled in
// global memory, B-panels staged in LDS). B staging uses the CDNA5 async
// global->LDS path (global_load_async_to_lds_b32 + s_wait_asynccnt), with LDS
// double-buffering in GEMM2 so the copy overlaps the WMMA stream.
// Workspace requirement: ~265 MB.
#include <hip/hip_runtime.h>
#include <stdint.h>

typedef __bf16 bf16;
typedef float v8f  __attribute__((ext_vector_type(8)));
typedef bf16  v16bf __attribute__((ext_vector_type(16)));

static __device__ __forceinline__ float lrelu(float v){ return v >= 0.f ? v : 0.01f * v; }

#define BN_   16
#define FIN_  128
#define FOUT_ 256
#define NPTS_ 2048
#define KNN_  20
#define GG_   512
#define CIN2_ 256          /* 2*FIN */
#define INNER2_ 2560       /* FIN*KNN */
#define EPS_  1e-5f

// ---------------------------------------------------------------- utilities
__global__ void k_zero(float* p, int n){
  int i = blockIdx.x * blockDim.x + threadIdx.x;
  if (i < n) p[i] = 0.f;
}

// x[b][c][n] -> xT[b][n][c] (f32) and xbT (bf16)
__global__ void k_prep_x(const float* __restrict__ x, float* __restrict__ xT, bf16* __restrict__ xbT){
  int e = blockIdx.x * 256 + threadIdx.x;          // B*FIN*NPTS = 4194304 exact
  int b = e >> 18;                                  // FIN*NPTS = 2^18
  int rem = e & ((1 << 18) - 1);
  int c = rem >> 11;                                // NPTS = 2^11
  int n = rem & (NPTS_ - 1);
  float v = x[e];
  size_t d = ((size_t)(b << 11) + n) * FIN_ + c;
  xT[d] = v;
  xbT[d] = (bf16)v;
}

// Pre-swizzle weights into WMMA A-fragment order (16-bit A 16x32 layout,
// contiguous k order): element (m,kk) of a tile -> lane = m + 16*(kk/16),
// half h = kk&15. Stored as [tile][lane][16 halfs] contiguous.
__global__ void k_prep_w(const float* __restrict__ w_inte, const float* __restrict__ w_conv2,
                         bf16* __restrict__ wA1, bf16* __restrict__ wA2){
  int d = blockIdx.x * 256 + threadIdx.x;           // 32768 + 655360 = 688128 exact
  if (d < 32768){
    int frag = d >> 9, rem = d & 511, lane = rem >> 4, h = rem & 15;
    int tm = frag >> 3, tk = frag & 7;
    int m = lane & 15;
    int kk = h + 16 * (lane >> 4);
    wA1[d] = (bf16)w_inte[(tm * 16 + m) * CIN2_ + tk * 32 + kk];
  } else {
    int d2 = d - 32768;
    int frag = d2 >> 9, rem = d2 & 511, lane = rem >> 4, h = rem & 15;
    int tm = frag / 80, tk = frag % 80;
    int m = lane & 15;
    int kk = h + 16 * (lane >> 4);
    int r = tk * 32 + kk;                           // r = c*KNN + k
    int c = r / KNN_, k = r % KNN_;
    wA2[d2] = (bf16)w_conv2[((size_t)(tm * 16 + m) * FIN_ + c) * KNN_ + k];
  }
}

// ------------------------------------------------------ kNN (top-21) per row
__global__ void k_topk(const float* __restrict__ xT, int* __restrict__ idxw){
  __shared__ float cent[FIN_];
  __shared__ float dist[NPTS_];
  __shared__ unsigned long long red[256];
  int n = blockIdx.x, b = blockIdx.y, tid = threadIdx.x;
  const float* base = xT + (size_t)b * NPTS_ * FIN_;
  if (tid < FIN_) cent[tid] = base[(size_t)n * FIN_ + tid];
  __syncthreads();
  for (int j = tid; j < NPTS_; j += 256){
    const float* pj = base + (size_t)j * FIN_;
    float d = 0.f;
    #pragma unroll 8
    for (int c = 0; c < FIN_; ++c){ float t = cent[c] - pj[c]; d += t * t; }
    dist[j] = d;
  }
  __syncthreads();
  for (int it = 0; it <= KNN_; ++it){               // rank 0 = self, skipped
    unsigned long long best = ~0ull;
    for (int j = tid; j < NPTS_; j += 256){
      unsigned long long key = ((unsigned long long)__float_as_uint(dist[j]) << 32) | (unsigned)j;
      best = key < best ? key : best;
    }
    red[tid] = best; __syncthreads();
    for (int s = 128; s > 0; s >>= 1){
      if (tid < s && red[tid + s] < red[tid]) red[tid] = red[tid + s];
      __syncthreads();
    }
    int jw = (int)(red[0] & 0xffffffffu);
    if (tid == 0){
      if (it > 0) idxw[(size_t)(b * NPTS_ + n) * KNN_ + (it - 1)] = jw;
      dist[jw] = __uint_as_float(0x7f800000u);      // +inf: remove winner
    }
    __syncthreads();
  }
}

// --------------------------------- GEMM1: h[b,o,n,k] = sum_c W1[o,c]*ee[c,n,k]
// A = W1 (128x256, pre-swizzled fragments from global). B panel = 128 edge
// columns x 256 inner, built in LDS. Slab 0 (central features, contiguous
// 256B/column) is staged with async global->LDS; slab 1 (nbr-central) needs
// VALU. Output: bf16 h in [b][n][c][k] plus per-channel BN sums.
__global__ void __launch_bounds__(256) k_gemm1(const bf16* __restrict__ xbT,
                                               const bf16* __restrict__ wA1,
                                               const int* __restrict__ idxw,
                                               bf16* __restrict__ hws,
                                               float* __restrict__ hsum, float* __restrict__ hsq){
  __shared__ __align__(32) bf16 Bp[128][144];       // 128 cols x 128-deep slab (+16 pad)
  __shared__ float ls[FIN_], lq[FIN_];
  int tid = threadIdx.x, lane = tid & 31, w = tid >> 5;
  int b = blockIdx.y;
  int pbase = blockIdx.x * 128;                     // flat (n,k) column base
  if (tid < FIN_){ ls[tid] = 0.f; lq[tid] = 0.f; }
  v8f acc[8];
  #pragma unroll
  for (int t = 0; t < 8; ++t)
    #pragma unroll
    for (int r = 0; r < 8; ++r) acc[t][r] = 0.f;
  const bf16* xb = xbT + (size_t)b * NPTS_ * FIN_;
  const int tm = w;                                 // 8 waves = 8 M-tiles (M=128)

  auto compute_slab = [&](int s){
    for (int tkk = 0; tkk < 4; ++tkk){
      int tk = s * 4 + tkk;
      v16bf afrag = *(const v16bf*)(wA1 + ((size_t)(tm * 8 + tk) * 32 + lane) * 16);
      int boff = tkk * 32 + ((lane >> 4) << 4);
      v16bf bcur = *(const v16bf*)(&Bp[(lane & 15)][boff]);
      #pragma unroll
      for (int tn = 0; tn < 8; ++tn){
        v16bf bnext = bcur;
        if (tn < 7) bnext = *(const v16bf*)(&Bp[(tn + 1) * 16 + (lane & 15)][boff]);
        acc[tn] = __builtin_amdgcn_wmma_f32_16x16x32_bf16(false, afrag, false, bcur,
                                                          (short)0, acc[tn], false, false);
        bcur = bnext;
      }
    }
  };

  // ---- slab 0: async global->LDS copy (central features)
  {
    uint32_t ldsbase = (uint32_t)(uintptr_t)(&Bp[0][0]);
    unsigned long long xbase = (unsigned long long)(uintptr_t)xb;
    for (int i = tid; i < 128 * 64; i += 256){      // 64 dwords = 128 halfs per column
      int col = i >> 6, r32 = i & 63;
      int n = (pbase + col) / KNN_;
      uint32_t voff = (uint32_t)n * (FIN_ * 2) + (uint32_t)(r32 * 4);
      uint32_t la = ldsbase + (uint32_t)(col * 288 + r32 * 4);
      asm volatile("global_load_async_to_lds_b32 %0, %1, %2"
                   :: "v"(la), "v"(voff), "s"(xbase) : "memory");
    }
    asm volatile("s_wait_asynccnt 0" ::: "memory");
    __syncthreads();
  }
  compute_slab(0);
  __syncthreads();

  // ---- slab 1: nbr - central (VALU staging)
  for (int i = tid; i < 128 * 128; i += 256){
    int col = i >> 7, rl = i & 127;
    int p = pbase + col;
    int n = p / KNN_, k = p % KNN_;
    int m = idxw[(size_t)(b * NPTS_ + n) * KNN_ + k];
    float v = (float)xb[(size_t)m * FIN_ + rl] - (float)xb[(size_t)n * FIN_ + rl];
    Bp[col][rl] = (bf16)v;
  }
  __syncthreads();
  compute_slab(1);
  __syncthreads();

  // ---- epilogue
  #pragma unroll
  for (int tn = 0; tn < 8; ++tn)
    #pragma unroll
    for (int r = 0; r < 8; ++r){
      float v = acc[tn][r];
      int o = tm * 16 + r + ((lane >> 4) << 3);     // C/D layout: row = r + 8*(lane/16)
      int p = pbase + tn * 16 + (lane & 15);
      int n = p / KNN_, k = p % KNN_;
      hws[((size_t)(b * NPTS_ + n) * FIN_ + o) * KNN_ + k] = (bf16)v;
      atomicAdd(&ls[o], v);
      atomicAdd(&lq[o], v * v);
    }
  __syncthreads();
  if (tid < FIN_){ atomicAdd(&hsum[tid], ls[tid]); atomicAdd(&hsq[tid], lq[tid]); }
}

__global__ void k_finalize(const float* __restrict__ sum, const float* __restrict__ sq,
                           float* __restrict__ mean, float* __restrict__ rstd,
                           int C, float inv_cnt){
  int i = blockIdx.x * blockDim.x + threadIdx.x;
  if (i < C){
    float m = sum[i] * inv_cnt;
    float v = sq[i] * inv_cnt - m * m;
    mean[i] = m; rstd[i] = rsqrtf(v + EPS_);
  }
}

// normalize + lrelu h in place (bf16); one (b,n,c) row of KNN_ elems per thread
__global__ void k_hnorm(bf16* __restrict__ hws, const float* __restrict__ mean,
                        const float* __restrict__ rstd, const float* __restrict__ g,
                        const float* __restrict__ be){
  const size_t rows = (size_t)BN_ * NPTS_ * FIN_;   // 4.19M rows of 20
  for (size_t row = (size_t)blockIdx.x * blockDim.x + threadIdx.x; row < rows;
       row += (size_t)gridDim.x * blockDim.x){
    int c = (int)(row & (FIN_ - 1));
    float m = mean[c], r = rstd[c], gg = g[c], bb = be[c];
    bf16* p = hws + row * KNN_;
    #pragma unroll
    for (int k = 0; k < KNN_; ++k){
      float v = (float)p[k];
      v = (v - m) * r * gg + bb;
      p[k] = (bf16)lrelu(v);
    }
  }
}

// --------------------------------- GEMM2: y[b,o,n] = sum_{c,k} W2[o,c,k]*hhat
// M=256 (16 M-tiles, 2/wave), 64 columns/block, inner 2560 in 20 slabs of 128.
// B slabs are contiguous in hws -> async global->LDS, double-buffered so the
// copy of slab s+1 overlaps the 32 WMMAs of slab s.
__global__ void __launch_bounds__(256) k_gemm2(const bf16* __restrict__ hws,
                                               const bf16* __restrict__ wA2,
                                               const float* __restrict__ bconv,
                                               float* __restrict__ yws,
                                               float* __restrict__ ysum, float* __restrict__ ysq){
  __shared__ __align__(32) bf16 Bp[2][64][144];     // double-buffered 64x128 slab (+16 pad)
  __shared__ float ls[FOUT_], lq[FOUT_];
  int tid = threadIdx.x, lane = tid & 31, w = tid >> 5;
  int b = blockIdx.y;
  int nbase = blockIdx.x * 64;
  if (tid < FOUT_){ ls[tid] = 0.f; lq[tid] = 0.f; }
  v8f acc[2][4];
  #pragma unroll
  for (int mi = 0; mi < 2; ++mi)
    #pragma unroll
    for (int tn = 0; tn < 4; ++tn)
      #pragma unroll
      for (int r = 0; r < 8; ++r) acc[mi][tn][r] = 0.f;

  uint32_t ldsbase[2] = { (uint32_t)(uintptr_t)(&Bp[0][0][0]),
                          (uint32_t)(uintptr_t)(&Bp[1][0][0]) };
  unsigned long long hbase = (unsigned long long)(uintptr_t)hws;

  auto stage_async = [&](int s, int buf){           // 64 cols x 256 bytes each
    for (int i = tid; i < 64 * 64; i += 256){
      int col = i >> 6, r32 = i & 63;
      int n = nbase + col;
      uint32_t voff = (uint32_t)(((size_t)(b * NPTS_ + n) * INNER2_ + (size_t)s * 128) * 2)
                    + (uint32_t)(r32 * 4);
      uint32_t la = ldsbase[buf] + (uint32_t)(col * 288 + r32 * 4);
      asm volatile("global_load_async_to_lds_b32 %0, %1, %2"
                   :: "v"(la), "v"(voff), "s"(hbase) : "memory");
    }
  };

  stage_async(0, 0);
  asm volatile("s_wait_asynccnt 0" ::: "memory");
  __syncthreads();

  for (int s = 0; s < 20; ++s){
    int buf = s & 1;
    if (s < 19) stage_async(s + 1, buf ^ 1);        // overlap copy with compute
    const bf16 (*__restrict__ Bs)[144] = Bp[buf];
    for (int tkk = 0; tkk < 4; ++tkk){
      int tk = s * 4 + tkk;
      v16bf a0 = *(const v16bf*)(wA2 + ((size_t)((w * 2 + 0) * 80 + tk) * 32 + lane) * 16);
      v16bf a1 = *(const v16bf*)(wA2 + ((size_t)((w * 2 + 1) * 80 + tk) * 32 + lane) * 16);
      int boff = tkk * 32 + ((lane >> 4) << 4);
      v16bf bcur = *(const v16bf*)(&Bs[(lane & 15)][boff]);
      #pragma unroll
      for (int tn = 0; tn < 4; ++tn){
        v16bf bnext = bcur;
        if (tn < 3) bnext = *(const v16bf*)(&Bs[(tn + 1) * 16 + (lane & 15)][boff]);
        acc[0][tn] = __builtin_amdgcn_wmma_f32_16x16x32_bf16(false, a0, false, bcur,
                                                             (short)0, acc[0][tn], false, false);
        acc[1][tn] = __builtin_amdgcn_wmma_f32_16x16x32_bf16(false, a1, false, bcur,
                                                             (short)0, acc[1][tn], false, false);
        bcur = bnext;
      }
    }
    asm volatile("s_wait_asynccnt 0" ::: "memory");
    __syncthreads();
  }

  #pragma unroll
  for (int mi = 0; mi < 2; ++mi)
    #pragma unroll
    for (int tn = 0; tn < 4; ++tn)
      #pragma unroll
      for (int r = 0; r < 8; ++r){
        int o = (w * 2 + mi) * 16 + r + ((lane >> 4) << 3);
        int col = nbase + tn * 16 + (lane & 15);
        float v = acc[mi][tn][r] + bconv[o];
        yws[((size_t)b * FOUT_ + o) * NPTS_ + col] = v;
        atomicAdd(&ls[o], v);
        atomicAdd(&lq[o], v * v);
      }
  __syncthreads();
  if (tid < FOUT_){ atomicAdd(&ysum[tid], ls[tid]); atomicAdd(&ysq[tid], lq[tid]); }
}

// BN1 + relu on y, accumulate stats of the result (for BN2)
__global__ void k_ynorm(const float* __restrict__ yws, float* __restrict__ y1,
                        const float* __restrict__ mean, const float* __restrict__ rstd,
                        const float* __restrict__ g, const float* __restrict__ be,
                        float* __restrict__ s1, float* __restrict__ q1){
  __shared__ float ls[256], lq[256];
  int bo = blockIdx.x;                              // b*256 + o
  int o = bo & 255;
  int tid = threadIdx.x;
  const float* row = yws + (size_t)bo * NPTS_;
  float* out = y1 + (size_t)bo * NPTS_;
  float m = mean[o], r = rstd[o], gg = g[o], bb = be[o];
  float acc = 0.f, accq = 0.f;
  for (int n = tid; n < NPTS_; n += 256){
    float v = (row[n] - m) * r * gg + bb;
    v = v > 0.f ? v : 0.f;                          // relu
    out[n] = v; acc += v; accq += v * v;
  }
  ls[tid] = acc; lq[tid] = accq; __syncthreads();
  for (int s = 128; s > 0; s >>= 1){
    if (tid < s){ ls[tid] += ls[tid + s]; lq[tid] += lq[tid + s]; }
    __syncthreads();
  }
  if (tid == 0){ atomicAdd(&s1[o], ls[0]); atomicAdd(&q1[o], lq[0]); }
}

// max over points per (b,c)
__global__ void k_xsmax(const float* __restrict__ x, float* __restrict__ xs0){
  __shared__ float lm[256];
  int bc = blockIdx.x, tid = threadIdx.x;
  const float* row = x + (size_t)bc * NPTS_;
  float m = -3.402823466e38f;
  for (int n = tid; n < NPTS_; n += 256) m = fmaxf(m, row[n]);
  lm[tid] = m; __syncthreads();
  for (int s = 128; s > 0; s >>= 1){
    if (tid < s) lm[tid] = fmaxf(lm[tid], lm[tid + s]);
    __syncthreads();
  }
  if (tid == 0) xs0[bc] = lm[0];
}

// tiny FC chain (fc1 -> bn -> lrelu -> fc2 -> bn -> lrelu -> g -> bn -> lrelu)
// biases omitted: they cancel exactly under the batch-dim BN that follows.
__global__ void __launch_bounds__(512) k_fc(const float* __restrict__ xs0,
    const float* __restrict__ w1, const float* __restrict__ g1, const float* __restrict__ be1,
    const float* __restrict__ w2, const float* __restrict__ g2, const float* __restrict__ be2,
    const float* __restrict__ wg, const float* __restrict__ ggm, const float* __restrict__ beg,
    float* __restrict__ xs2, float* __restrict__ gout){
  __shared__ float s0[BN_ * FIN_];
  __shared__ float t1[BN_ * FIN_];
  __shared__ float t2[BN_ * FOUT_];
  int tid = threadIdx.x;
  for (int i = tid; i < BN_ * FIN_; i += 512) s0[i] = xs0[i];
  __syncthreads();
  for (int u = tid; u < BN_ * FIN_; u += 512){
    int bb = u >> 7, j = u & 127; float a = 0.f;
    for (int c = 0; c < FIN_; ++c) a += s0[bb * FIN_ + c] * w1[j * FIN_ + c];
    t1[u] = a;
  }
  __syncthreads();
  if (tid < FIN_){
    float m = 0.f, q = 0.f;
    for (int bb = 0; bb < BN_; ++bb){ float v = t1[bb * FIN_ + tid]; m += v; q += v * v; }
    m *= (1.f / BN_); q = q * (1.f / BN_) - m * m;
    float r = rsqrtf(q + EPS_);
    for (int bb = 0; bb < BN_; ++bb)
      t1[bb * FIN_ + tid] = lrelu((t1[bb * FIN_ + tid] - m) * r * g1[tid] + be1[tid]);
  }
  __syncthreads();
  for (int u = tid; u < BN_ * FOUT_; u += 512){
    int bb = u >> 8, j = u & 255; float a = 0.f;
    for (int c = 0; c < FIN_; ++c) a += t1[bb * FIN_ + c] * w2[j * FIN_ + c];
    t2[u] = a;
  }
  __syncthreads();
  if (tid < FOUT_){
    float m = 0.f, q = 0.f;
    for (int bb = 0; bb < BN_; ++bb){ float v = t2[bb * FOUT_ + tid]; m += v; q += v * v; }
    m *= (1.f / BN_); q = q * (1.f / BN_) - m * m;
    float r = rsqrtf(q + EPS_);
    for (int bb = 0; bb < BN_; ++bb){
      float v = lrelu((t2[bb * FOUT_ + tid] - m) * r * g2[tid] + be2[tid]);
      t2[bb * FOUT_ + tid] = v;
      xs2[bb * FOUT_ + tid] = v;
    }
  }
  __syncthreads();
  for (int u = tid; u < BN_ * GG_; u += 512){
    int bb = u >> 9, j = u & 511; float a = 0.f;
    for (int c = 0; c < FOUT_; ++c) a += t2[bb * FOUT_ + c] * wg[j * FOUT_ + c];
    gout[u] = a;                                    // raw, normalized below
  }
  __syncthreads();
  if (tid < GG_){
    float m = 0.f, q = 0.f;
    for (int bb = 0; bb < BN_; ++bb){ float v = gout[bb * GG_ + tid]; m += v; q += v * v; }
    m *= (1.f / BN_); q = q * (1.f / BN_) - m * m;
    float r = rsqrtf(q + EPS_);
    for (int bb = 0; bb < BN_; ++bb)
      gout[bb * GG_ + tid] = lrelu((gout[bb * GG_ + tid] - m) * r * ggm[tid] + beg[tid]);
  }
}

// final assembly: x_out (B,512,N) then g_out (B,768,N), x_ec computed on the fly
__global__ void k_out(const float* __restrict__ y1, const float* __restrict__ y1m,
                      const float* __restrict__ y1r, const float* __restrict__ gbn,
                      const float* __restrict__ bebn, const float* __restrict__ xs2,
                      const float* __restrict__ gvec, float* __restrict__ out){
  const size_t X0 = (size_t)BN_ * 512 * NPTS_;
  const size_t TOT = X0 + (size_t)BN_ * 768 * NPTS_;
  for (size_t e = (size_t)blockIdx.x * blockDim.x + threadIdx.x; e < TOT;
       e += (size_t)gridDim.x * blockDim.x){
    float v;
    if (e < X0){
      size_t rem = e;
      int b = (int)(rem / ((size_t)512 * NPTS_)); rem -= (size_t)b * 512 * NPTS_;
      int j = (int)(rem / NPTS_); int n = (int)(rem - (size_t)j * NPTS_);
      if (j < FOUT_) v = xs2[b * FOUT_ + j];
      else {
        int o = j - FOUT_;
        v = lrelu((y1[((size_t)b * FOUT_ + o) * NPTS_ + n] - y1m[o]) * y1r[o] * gbn[o] + bebn[o]);
      }
    } else {
      size_t rem = e - X0;
      int b = (int)(rem / ((size_t)768 * NPTS_)); rem -= (size_t)b * 768 * NPTS_;
      int j = (int)(rem / NPTS_); int n = (int)(rem - (size_t)j * NPTS_);
      if (j < GG_) v = gvec[b * GG_ + j];
      else {
        int o = j - GG_;
        v = lrelu((y1[((size_t)b * FOUT_ + o) * NPTS_ + n] - y1m[o]) * y1r[o] * gbn[o] + bebn[o]);
      }
    }
    out[e] = v;
  }
}

// ---------------------------------------------------------------- launcher
extern "C" void kernel_launch(void* const* d_in, const int* in_sizes, int n_in,
                              void* d_out, int out_size, void* d_ws, size_t ws_size,
                              hipStream_t stream){
  const float* x       = (const float*)d_in[0];
  const float* w_inte  = (const float*)d_in[2];
  const float* g_inte  = (const float*)d_in[4];
  const float* be_inte = (const float*)d_in[5];
  const float* w_conv2 = (const float*)d_in[6];
  const float* b_conv2 = (const float*)d_in[7];
  const float* g_conv2 = (const float*)d_in[8];
  const float* be_conv2= (const float*)d_in[9];
  const float* g_bn    = (const float*)d_in[10];
  const float* be_bn   = (const float*)d_in[11];
  const float* w_fc1   = (const float*)d_in[12];
  const float* g_fc1   = (const float*)d_in[14];
  const float* be_fc1  = (const float*)d_in[15];
  const float* w_fc2   = (const float*)d_in[16];
  const float* g_fc2   = (const float*)d_in[18];
  const float* be_fc2  = (const float*)d_in[19];
  const float* w_g     = (const float*)d_in[20];
  const float* g_g     = (const float*)d_in[22];
  const float* be_g    = (const float*)d_in[23];

  char* ws = (char*)d_ws;
  size_t off = 0;
  auto alloc = [&](size_t bytes) -> char* {
    char* p = ws + off;
    off = (off + bytes + 255) & ~(size_t)255;
    return p;
  };
  float* xT   = (float*)alloc((size_t)BN_ * NPTS_ * FIN_ * 4);          // 16 MB
  bf16*  xbT  = (bf16*) alloc((size_t)BN_ * NPTS_ * FIN_ * 2);          // 8 MB
  bf16*  wA1  = (bf16*) alloc((size_t)FIN_ * CIN2_ * 2);                // 64 KB
  bf16*  wA2  = (bf16*) alloc((size_t)FOUT_ * INNER2_ * 2);             // 1.3 MB
  int*   idxw = (int*)  alloc((size_t)BN_ * NPTS_ * KNN_ * 4);          // 2.6 MB
  bf16*  hws  = (bf16*) alloc((size_t)BN_ * NPTS_ * INNER2_ * 2);       // 168 MB
  float* yws  = (float*)alloc((size_t)BN_ * FOUT_ * NPTS_ * 4);         // 33.5 MB
  float* y1   = (float*)alloc((size_t)BN_ * FOUT_ * NPTS_ * 4);         // 33.5 MB
  float* xs0  = (float*)alloc((size_t)BN_ * FIN_ * 4);
  float* xs2  = (float*)alloc((size_t)BN_ * FOUT_ * 4);
  float* gws  = (float*)alloc((size_t)BN_ * GG_ * 4);
  float* stats= (float*)alloc(4096 * 4);
  (void)ws_size; (void)in_sizes; (void)n_in; (void)out_size;

  float* hsum = stats +    0; float* hsq  = stats +  128;
  float* hmean= stats +  256; float* hrstd= stats +  384;
  float* ysum = stats +  512; float* ysq  = stats +  768;
  float* ymean= stats + 1024; float* yrstd= stats + 1280;
  float* y1sum= stats + 1536; float* y1sq = stats + 1792;
  float* y1mn = stats + 2048; float* y1rs = stats + 2304;

  k_zero  <<<16, 256, 0, stream>>>(stats, 4096);
  k_prep_x<<<16384, 256, 0, stream>>>(x, xT, xbT);
  k_prep_w<<<2688, 256, 0, stream>>>(w_inte, w_conv2, wA1, wA2);
  k_topk  <<<dim3(NPTS_, BN_), 256, 0, stream>>>(xT, idxw);
  k_gemm1 <<<dim3(320, BN_), 256, 0, stream>>>(xbT, wA1, idxw, hws, hsum, hsq);
  k_finalize<<<1, 128, 0, stream>>>(hsum, hsq, hmean, hrstd, FIN_,
                                    1.f / ((float)BN_ * NPTS_ * KNN_));
  k_hnorm <<<8192, 256, 0, stream>>>(hws, hmean, hrstd, g_inte, be_inte);
  k_gemm2 <<<dim3(32, BN_), 256, 0, stream>>>(hws, wA2, b_conv2, yws, ysum, ysq);
  k_finalize<<<1, 256, 0, stream>>>(ysum, ysq, ymean, yrstd, FOUT_,
                                    1.f / ((float)BN_ * NPTS_));
  k_ynorm <<<BN_ * FOUT_, 256, 0, stream>>>(yws, y1, ymean, yrstd, g_conv2, be_conv2, y1sum, y1sq);
  k_finalize<<<1, 256, 0, stream>>>(y1sum, y1sq, y1mn, y1rs, FOUT_,
                                    1.f / ((float)BN_ * NPTS_));
  k_xsmax <<<BN_ * FIN_, 256, 0, stream>>>(x, xs0);
  k_fc    <<<1, 512, 0, stream>>>(xs0, w_fc1, g_fc1, be_fc1, w_fc2, g_fc2, be_fc2,
                                  w_g, g_g, be_g, xs2, gws);
  k_out   <<<20480, 256, 0, stream>>>(y1, y1mn, y1rs, g_bn, be_bn, xs2, gws, (float*)d_out);
}
